// InputAttentionModule_21930103013920
// MI455X (gfx1250) — matrix operand order repaired and checked by reading, not compiled, not run on hardware
//
#include <hip/hip_runtime.h>
#include <cmath>

typedef __attribute__((ext_vector_type(16))) _Float16 v16h;
typedef __attribute__((ext_vector_type(8)))  float    v8f;

#define B_     32
#define N_     16384
#define IND    128
#define DIM    2
#define SLOTS  10
#define HID    10
#define ITERS  3
#define LN_EPS 1e-3f
#define EPS_   1e-8f
#define SCALE_ 0.70710678118654752f   // DIM^-0.5

__device__ __forceinline__ float sigm(float x) { return 1.0f / (1.0f + __expf(-x)); }

// ---------------------------------------------------------------------------
// Kernel 0: latent = mu + exp(log_sigma)*noise ; q = LN(latent)@Wq^T ; upd = 0
// ---------------------------------------------------------------------------
__global__ void k_init(const float* __restrict__ noise, const float* __restrict__ mu,
                       const float* __restrict__ ls, const float* __restrict__ Wq,
                       const float* __restrict__ lw, const float* __restrict__ lb,
                       float* __restrict__ latent, float* __restrict__ q,
                       float* __restrict__ upd) {
    int t = threadIdx.x;
    if (t >= B_ * SLOTS) return;
    int i = t % SLOTS;
    int base = t * DIM;
    int p = i * DIM;
    float l0 = mu[p + 0] + __expf(ls[p + 0]) * noise[base + 0];
    float l1 = mu[p + 1] + __expf(ls[p + 1]) * noise[base + 1];
    latent[base + 0] = l0;
    latent[base + 1] = l1;
    upd[base + 0] = 0.0f;
    upd[base + 1] = 0.0f;
    float m  = 0.5f * (l0 + l1);
    float vv = 0.5f * ((l0 - m) * (l0 - m) + (l1 - m) * (l1 - m));
    float iq = rsqrtf(vv + LN_EPS);
    float n0 = (l0 - m) * iq * lw[0] + lb[0];
    float n1 = (l1 - m) * iq * lw[1] + lb[1];
    q[base + 0] = n0 * Wq[0] + n1 * Wq[1];
    q[base + 1] = n0 * Wq[2] + n1 * Wq[3];
}

// ---------------------------------------------------------------------------
// Kernel 1: fused LN(inputs) + [k|v] projection via v_wmma_f32_16x16x32_f16.
// B panel is staged DENSE and zero-padded (128x16, cols 0..3 live) so the
// fragment gather is pure ds_load_b128 with no per-lane cndmask on the
// WMMA critical path. kv packed as [B*N][4] = (k0,k1,v0,v1).
// ---------------------------------------------------------------------------
__global__ __launch_bounds__(128)
void k_project(const float* __restrict__ x, const float* __restrict__ Wk,
               const float* __restrict__ Wv, const float* __restrict__ lnw,
               const float* __restrict__ lnb, float* __restrict__ kv) {
    __shared__ _Float16 xt[4][16][IND];   // per-wave 16x128 f16 A tiles (16 KB)
    __shared__ _Float16 bw[16][IND];      // dense zero-padded B panel (4 KB)

    const int tid  = threadIdx.x;
    const int w    = tid >> 5;
    const int lane = tid & 31;

    // Stage dense B panel: column c, row k. Columns 4..15 are zeros.
    for (int i = tid; i < 16 * IND; i += 128) {
        int c = i >> 7, k = i & 127;
        float val = (c == 0) ? Wk[k] : (c == 1) ? Wk[IND + k]
                  : (c == 2) ? Wv[k] : (c == 3) ? Wv[IND + k] : 0.0f;
        bw[c][k] = (_Float16)val;
    }

    const float4 lw = reinterpret_cast<const float4*>(lnw)[lane];
    const float4 lb = reinterpret_cast<const float4*>(lnb)[lane];
    const long row0 = (long)blockIdx.x * 64 + (long)w * 16;

    // LayerNorm 16 rows cooperatively (32 lanes x float4 = 128 cols), f16 to LDS.
    for (int rr = 0; rr < 16; ++rr) {
        long row = row0 + rr;
        float4 xv = reinterpret_cast<const float4*>(x + row * IND)[lane];
        float s  = xv.x + xv.y + xv.z + xv.w;
        float sq = xv.x * xv.x + xv.y * xv.y + xv.z * xv.z + xv.w * xv.w;
        #pragma unroll
        for (int m = 16; m >= 1; m >>= 1) {
            s  += __shfl_xor(s,  m, 32);
            sq += __shfl_xor(sq, m, 32);
        }
        float mean = s * (1.0f / 128.0f);
        float var  = sq * (1.0f / 128.0f) - mean * mean;
        float inv  = rsqrtf(var + LN_EPS);
        xt[w][rr][lane * 4 + 0] = (_Float16)((xv.x - mean) * inv * lw.x + lb.x);
        xt[w][rr][lane * 4 + 1] = (_Float16)((xv.y - mean) * inv * lw.y + lb.y);
        xt[w][rr][lane * 4 + 2] = (_Float16)((xv.z - mean) * inv * lw.z + lb.z);
        xt[w][rr][lane * 4 + 3] = (_Float16)((xv.w - mean) * inv * lw.w + lb.w);
    }
    __syncthreads();

    // WMMA fragment gather: ISA 7.12.2 wave32 layouts.
    const int arow  = lane & 15;
    const int abase = (lane < 16) ? 0 : 8;    // A: K sub-offset per lane half
    const int bcol  = lane & 15;              // B: column = lane % 16
    const int bkb   = (lane < 16) ? 0 : 16;   // unused; B reuses A's koff (0/8 per lane half)

    v8f acc = {};
    #pragma unroll
    for (int kk = 0; kk < 4; ++kk) {
        const int K0 = kk * 32;
        v16h a, bm;
        #pragma unroll
        for (int v = 0; v < 8; ++v) {
            int koff = abase + ((v < 4) ? (2 * v) : (16 + 2 * (v - 4)));
            a[2 * v]     = xt[w][arow][K0 + koff];
            a[2 * v + 1] = xt[w][arow][K0 + koff + 1];
            int kb = K0 + koff;
            bm[2 * v]     = bw[bcol][kb];
            bm[2 * v + 1] = bw[bcol][kb + 1];
        }
        acc = __builtin_amdgcn_wmma_f32_16x16x32_f16(
            /*neg_a=*/false, a, /*neg_b=*/false, bm,
            /*c_mod=*/(short)0, acc, /*reuse_a=*/false, /*reuse_b=*/false);
    }

    // C layout: VGPR r -> (M=r, N=lane) lanes 0-15 ; (M=r+8, N=lane-16) lanes 16-31.
    // Columns N=0..3 hold (k0,k1,v0,v1).
    if (lane < 4 || (lane >= 16 && lane < 20)) {
        int comp = lane & 3;
        int roff = (lane < 4) ? 0 : 8;
        #pragma unroll
        for (int r = 0; r < 8; ++r) {
            long row = row0 + roff + r;
            kv[row * 4 + comp] = acc[r];
        }
    }
}

// ---------------------------------------------------------------------------
// Kernel 2: per j: dots over 10 slots, softmax across slots, accumulate
// attn*v into upd[b][10][2]. Wave shuffle reduce -> LDS atomics -> global.
// ---------------------------------------------------------------------------
__global__ __launch_bounds__(256)
void k_attend(const float* __restrict__ kv, const float* __restrict__ q,
              float* __restrict__ upd) {
    __shared__ float qs[SLOTS * DIM];
    __shared__ float accs[SLOTS * DIM];
    const int b   = blockIdx.x >> 6;       // 64 blocks of 256 threads per batch
    const int jb  = blockIdx.x & 63;
    const int tid = threadIdx.x;
    if (tid < SLOTS * DIM) {
        qs[tid]   = q[b * SLOTS * DIM + tid];
        accs[tid] = 0.0f;
    }
    __syncthreads();

    const long j = (long)jb * 256 + tid;
    const long idx = (long)b * N_ + j;
    float4 kvv = reinterpret_cast<const float4*>(kv)[idx];
    if (j + 4096 < N_)   // speculative L2 prefetch of upcoming k/v
        __builtin_prefetch(kv + (idx + 4096) * 4, 0, 0);

    float d[SLOTS];
    float mx = -1e30f;
    #pragma unroll
    for (int i = 0; i < SLOTS; ++i) {
        d[i] = SCALE_ * (qs[i * 2] * kvv.x + qs[i * 2 + 1] * kvv.y);
        mx = fmaxf(mx, d[i]);
    }
    float s = 0.0f;
    #pragma unroll
    for (int i = 0; i < SLOTS; ++i) { d[i] = __expf(d[i] - mx); s += d[i]; }
    const float inv = 1.0f / s;

    #pragma unroll
    for (int i = 0; i < SLOTS; ++i) {
        float a  = d[i] * inv + EPS_;
        float u0 = a * kvv.z;
        float u1 = a * kvv.w;
        #pragma unroll
        for (int m = 16; m >= 1; m >>= 1) {
            u0 += __shfl_xor(u0, m, 32);
            u1 += __shfl_xor(u1, m, 32);
        }
        if ((tid & 31) == 0) {
            atomicAdd(&accs[i * 2 + 0], u0);
            atomicAdd(&accs[i * 2 + 1], u1);
        }
    }
    __syncthreads();
    if (tid < SLOTS * DIM) atomicAdd(&upd[b * SLOTS * DIM + tid], accs[tid]);
}

// ---------------------------------------------------------------------------
// Kernel 3: GRU cell + LN + MLP residual per (b,slot); emits next q or output.
// Also re-zeros upd for the next iteration.
// ---------------------------------------------------------------------------
__global__ void k_update(const float* __restrict__ Wih, const float* __restrict__ Whh,
                         const float* __restrict__ bih, const float* __restrict__ bhh,
                         const float* __restrict__ W1,  const float* __restrict__ b1,
                         const float* __restrict__ W2,  const float* __restrict__ b2,
                         const float* __restrict__ Wq,
                         const float* __restrict__ llw, const float* __restrict__ llb,
                         const float* __restrict__ lfw, const float* __restrict__ lfb,
                         float* __restrict__ latent, float* __restrict__ q,
                         float* __restrict__ upd, float* __restrict__ out, int last) {
    int t = threadIdx.x;
    if (t >= B_ * SLOTS) return;
    int base = t * DIM;

    float u0 = upd[base + 0], u1 = upd[base + 1];
    upd[base + 0] = 0.0f;
    upd[base + 1] = 0.0f;
    float l0 = latent[base + 0], l1 = latent[base + 1];

    float gi[6], gh[6];
    #pragma unroll
    for (int g = 0; g < 6; ++g) {
        gi[g] = Wih[g * 2] * u0 + Wih[g * 2 + 1] * u1 + bih[g];
        gh[g] = Whh[g * 2] * l0 + Whh[g * 2 + 1] * l1 + bhh[g];
    }
    float r0 = sigm(gi[0] + gh[0]), r1 = sigm(gi[1] + gh[1]);
    float z0 = sigm(gi[2] + gh[2]), z1 = sigm(gi[3] + gh[3]);
    float n0 = tanhf(gi[4] + r0 * gh[4]);
    float n1 = tanhf(gi[5] + r1 * gh[5]);
    float h0 = (1.0f - z0) * n0 + z0 * l0;
    float h1 = (1.0f - z1) * n1 + z1 * l1;

    // LN_ff + MLP residual
    float m  = 0.5f * (h0 + h1);
    float vv = 0.5f * ((h0 - m) * (h0 - m) + (h1 - m) * (h1 - m));
    float iq = rsqrtf(vv + LN_EPS);
    float f0 = (h0 - m) * iq * lfw[0] + lfb[0];
    float f1 = (h1 - m) * iq * lfw[1] + lfb[1];
    float o0 = b2[0], o1 = b2[1];
    #pragma unroll
    for (int jj = 0; jj < HID; ++jj) {
        float hid = fmaxf(0.0f, f0 * W1[jj * 2] + f1 * W1[jj * 2 + 1] + b1[jj]);
        o0 += hid * W2[jj];         // W2[0][jj]
        o1 += hid * W2[HID + jj];   // W2[1][jj]
    }
    float L0 = h0 + o0, L1 = h1 + o1;
    latent[base + 0] = L0;
    latent[base + 1] = L1;

    if (last) {
        out[base + 0] = L0;
        out[base + 1] = L1;
    } else {
        float mm = 0.5f * (L0 + L1);
        float v2 = 0.5f * ((L0 - mm) * (L0 - mm) + (L1 - mm) * (L1 - mm));
        float i2 = rsqrtf(v2 + LN_EPS);
        float a0 = (L0 - mm) * i2 * llw[0] + llb[0];
        float a1 = (L1 - mm) * i2 * llw[1] + llb[1];
        q[base + 0] = a0 * Wq[0] + a1 * Wq[1];
        q[base + 1] = a0 * Wq[2] + a1 * Wq[3];
    }
}

// ---------------------------------------------------------------------------
extern "C" void kernel_launch(void* const* d_in, const int* in_sizes, int n_in,
                              void* d_out, int out_size, void* d_ws, size_t ws_size,
                              hipStream_t stream) {
    const float* inputs = (const float*)d_in[0];
    const float* noise  = (const float*)d_in[1];
    const float* mu     = (const float*)d_in[2];
    const float* ls     = (const float*)d_in[3];
    const float* Wq     = (const float*)d_in[4];
    const float* Wk     = (const float*)d_in[5];
    const float* Wv     = (const float*)d_in[6];
    const float* W1     = (const float*)d_in[7];
    const float* b1     = (const float*)d_in[8];
    const float* W2     = (const float*)d_in[9];
    const float* b2     = (const float*)d_in[10];
    const float* Wih    = (const float*)d_in[11];
    const float* Whh    = (const float*)d_in[12];
    const float* bih    = (const float*)d_in[13];
    const float* bhh    = (const float*)d_in[14];
    const float* lin_w  = (const float*)d_in[15];
    const float* lin_b  = (const float*)d_in[16];
    const float* llat_w = (const float*)d_in[17];
    const float* llat_b = (const float*)d_in[18];
    const float* lff_w  = (const float*)d_in[19];
    const float* lff_b  = (const float*)d_in[20];
    float* out = (float*)d_out;

    float* kv     = (float*)d_ws;                    // [B*N][4]  (8 MB)
    float* latent = kv + (size_t)B_ * N_ * 4;        // [B][10][2]
    float* q      = latent + B_ * SLOTS * DIM;       // [B][10][2]
    float* upd    = q + B_ * SLOTS * DIM;            // [B][10][2] accumulator

    k_init<<<1, 320, 0, stream>>>(noise, mu, ls, Wq, llat_w, llat_b, latent, q, upd);
    k_project<<<(B_ * N_) / 64, 128, 0, stream>>>(inputs, Wk, Wv, lin_w, lin_b, kv);
    for (int it = 0; it < ITERS; ++it) {
        k_attend<<<B_ * (N_ / 256), 256, 0, stream>>>(kv, q, upd);
        k_update<<<1, 320, 0, stream>>>(Wih, Whh, bih, bhh, W1, b1, W2, b2, Wq,
                                        llat_w, llat_b, lff_w, lff_b,
                                        latent, q, upd, out,
                                        (it == ITERS - 1) ? 1 : 0);
    }
}
